// LSTMLayer_35330400977123
// MI455X (gfx1250) — compile-verified, matrix operand order checked by reading, not run on hardware
//
#include <hip/hip_runtime.h>
#include <hip/hip_bf16.h>
#include <math.h>

#define HID 1024
#define INSZ 1024
#define BATCH 64
#define SEQ 512
#define G4 4096

typedef __bf16 bf16t;
typedef __attribute__((ext_vector_type(16))) __bf16 v16bf;
typedef __attribute__((ext_vector_type(8)))  __bf16 v8bf;
typedef __attribute__((ext_vector_type(8)))  float  v8f;

static __device__ __forceinline__ bf16t f2bf(float f) {
    union { float f; unsigned u; } a; a.f = f;
    unsigned r = a.u + 0x7FFFu + ((a.u >> 16) & 1u);
    unsigned short s = (unsigned short)(r >> 16);
    union { unsigned short u; bf16t b; } o; o.u = s;
    return o.b;
}

// Load one lane's 16-element bf16 WMMA fragment: elements [0..7] at p, [8..15] at p+16
// (ISA 7.12.2: 16-bit A 16x32 layout; B stored transposed uses the same pattern).
static __device__ __forceinline__ v16bf load_frag(const bf16t* p) {
    v8bf lo = *reinterpret_cast<const v8bf*>(p);
    v8bf hi = *reinterpret_cast<const v8bf*>(p + 16);
    return __builtin_shufflevector(lo, hi, 0,1,2,3,4,5,6,7,8,9,10,11,12,13,14,15);
}

static __device__ __forceinline__ float sigm(float x) {
    return 1.0f / (1.0f + __expf(-x));
}

// ---- Pack W_{f,i,c,o}[2048,1024] fp32 -> WxT[4096][1024], WhT[4096][1024] bf16 (transposed) ----
__global__ void pack_weights(const float* __restrict__ Wf, const float* __restrict__ Wi,
                             const float* __restrict__ Wc, const float* __restrict__ Wo,
                             bf16t* __restrict__ WxT, bf16t* __restrict__ WhT) {
    unsigned idx = blockIdx.x * 256u + threadIdx.x;      // 0 .. 2*4096*1024
    unsigned matId = idx >> 22;                          // 0 = Wx part, 1 = Wh part
    unsigned rem   = idx & ((1u << 22) - 1u);
    unsigned row   = rem >> 10;                          // gate*H + n  (0..4095)
    unsigned k     = rem & 1023u;
    unsigned gate  = row >> 10;
    unsigned n     = row & 1023u;
    const float* W = (gate == 0) ? Wf : (gate == 1) ? Wi : (gate == 2) ? Wc : Wo;
    float v = W[(size_t)(k + (matId ? INSZ : 0)) * HID + n];
    (matId ? WhT : WxT)[(size_t)row * 1024 + k] = f2bf(v);
}

__global__ void pack_bias(const float* __restrict__ bf_, const float* __restrict__ bi_,
                          const float* __restrict__ bc_, const float* __restrict__ bo_,
                          float* __restrict__ bias) {
    int i = blockIdx.x * 256 + threadIdx.x;              // 0..4095
    int gate = i >> 10, n = i & 1023;
    const float* b = (gate == 0) ? bf_ : (gate == 1) ? bi_ : (gate == 2) ? bc_ : bo_;
    bias[i] = b[n];
}

__global__ void convert_x(const float* __restrict__ x, bf16t* __restrict__ xb, int n) {
    int i = (blockIdx.x * 256 + threadIdx.x) * 4;
    if (i + 3 < n) {
        xb[i + 0] = f2bf(x[i + 0]);
        xb[i + 1] = f2bf(x[i + 1]);
        xb[i + 2] = f2bf(x[i + 2]);
        xb[i + 3] = f2bf(x[i + 3]);
    }
}

__global__ void init_state(bf16t* __restrict__ h0, bf16t* __restrict__ h1,
                           float* __restrict__ c) {
    int i = blockIdx.x * 256 + threadIdx.x;              // 0..65535
    union { unsigned short u; bf16t b; } z; z.u = 0;
    h0[i] = z.b; h1[i] = z.b; c[i] = 0.0f;
}

// ---- Phase 1: xg[32768][4096] = x_bf16[32768][1024] @ WxT^T + bias (bf16 WMMA, f32 acc) ----
// Register-blocked 1 M-tile x 4 N-tiles per wave, software-pipelined one iteration deep.
// __launch_bounds__(256, 1): lift the default occupancy-driven VGPR cap so the ~130 live
// registers (4 acc + current/next fragment sets) stay in VGPRs instead of scratch spills.
// 2048 M-tiles x 64 N-groups = 131072 waves -> 16384 blocks of 8 waves.
__global__ __launch_bounds__(256, 1)
void gemm_xproj(const bf16t* __restrict__ xb, const bf16t* __restrict__ WxT,
                const float* __restrict__ bias, float* __restrict__ xg) {
    const int lane = threadIdx.x & 31;
    const int wave = threadIdx.x >> 5;
    const unsigned wid = blockIdx.x * 8u + wave;
    const unsigned ng  = wid & 63u;                      // N-group (4 tiles each)
    const unsigned mt  = wid >> 6;                       // M tile (0..2047)
    const int lo = lane & 15, khi = lane >> 4;

    const bf16t* arow  = xb  + (size_t)(mt * 16 + lo) * 1024 + 8 * khi;
    const bf16t* brow0 = WxT + (size_t)((ng * 4 + 0) * 16 + lo) * 1024 + 8 * khi;
    const bf16t* brow1 = WxT + (size_t)((ng * 4 + 1) * 16 + lo) * 1024 + 8 * khi;
    const bf16t* brow2 = WxT + (size_t)((ng * 4 + 2) * 16 + lo) * 1024 + 8 * khi;
    const bf16t* brow3 = WxT + (size_t)((ng * 4 + 3) * 16 + lo) * 1024 + 8 * khi;

    v8f acc0 = {}, acc1 = {}, acc2 = {}, acc3 = {};
    // prologue
    v16bf a  = load_frag(arow);
    v16bf b0 = load_frag(brow0);
    v16bf b1 = load_frag(brow1);
    v16bf b2 = load_frag(brow2);
    v16bf b3 = load_frag(brow3);
    #pragma unroll 2
    for (int k = 0; k < 992; k += 32) {
        __builtin_prefetch(arow + k + 288, 0, 1);
        v16bf an  = load_frag(arow  + k + 32);
        v16bf b0n = load_frag(brow0 + k + 32);
        v16bf b1n = load_frag(brow1 + k + 32);
        v16bf b2n = load_frag(brow2 + k + 32);
        v16bf b3n = load_frag(brow3 + k + 32);
        acc0 = __builtin_amdgcn_wmma_f32_16x16x32_bf16(false, a, false, b0, (short)0, acc0, false, false);
        acc1 = __builtin_amdgcn_wmma_f32_16x16x32_bf16(false, a, false, b1, (short)0, acc1, false, false);
        acc2 = __builtin_amdgcn_wmma_f32_16x16x32_bf16(false, a, false, b2, (short)0, acc2, false, false);
        acc3 = __builtin_amdgcn_wmma_f32_16x16x32_bf16(false, a, false, b3, (short)0, acc3, false, false);
        a = an; b0 = b0n; b1 = b1n; b2 = b2n; b3 = b3n;
    }
    // epilogue
    acc0 = __builtin_amdgcn_wmma_f32_16x16x32_bf16(false, a, false, b0, (short)0, acc0, false, false);
    acc1 = __builtin_amdgcn_wmma_f32_16x16x32_bf16(false, a, false, b1, (short)0, acc1, false, false);
    acc2 = __builtin_amdgcn_wmma_f32_16x16x32_bf16(false, a, false, b2, (short)0, acc2, false, false);
    acc3 = __builtin_amdgcn_wmma_f32_16x16x32_bf16(false, a, false, b3, (short)0, acc3, false, false);

    #pragma unroll
    for (int j = 0; j < 4; ++j) {
        v8f acc = (j == 0) ? acc0 : (j == 1) ? acc1 : (j == 2) ? acc2 : acc3;
        const unsigned col = (ng * 4 + j) * 16 + lo;
        const float bv = bias[col];
        #pragma unroll
        for (int i = 0; i < 8; ++i) {
            unsigned m = mt * 16 + i + 8 * khi;          // C layout: VGPR i = row i + 8*(lane>>4)
            xg[(size_t)m * G4 + col] = acc[i] + bv;
        }
    }
}

// ---- Phase 2: one kernel per timestep: g = xg[s] + h @ Wh, gates, state update ----
// grid (64 hidden-chunks, 4 M-tiles), 128 threads = 4 waves = one wave per gate.
// Two interleaved K-accumulators break the WMMA->WMMA RAW chain; explicit pipelining
// keeps next iteration's 4 fragments in flight during the current WMMA pair.
// __launch_bounds__(128, 1): full VGPR budget, no spills on the sequential critical path.
__global__ __launch_bounds__(128, 1)
void lstm_step(const bf16t* __restrict__ WhT, const float* __restrict__ xg,
               const bf16t* __restrict__ h_in, bf16t* __restrict__ h_out,
               float* __restrict__ c, float* __restrict__ hs,
               float* __restrict__ cs, int s) {
    __shared__ float g_sm[4][16][16];
    const int lane = threadIdx.x & 31;
    const int gate = threadIdx.x >> 5;                   // 0..3: f, i, c~, o
    const int hc   = blockIdx.x;                         // hidden chunk 0..63
    const int mt   = blockIdx.y;                         // batch tile 0..3
    const int lo = lane & 15, khi = lane >> 4;

    const bf16t* arow = h_in + (size_t)(mt * 16 + lo) * 1024 + 8 * khi;
    const bf16t* brow = WhT  + (size_t)(gate * 1024 + hc * 16 + lo) * 1024 + 8 * khi;

    v8f acc0 = {}, acc1 = {};
    // prologue
    v16bf a0 = load_frag(arow);
    v16bf b0 = load_frag(brow);
    v16bf a1 = load_frag(arow + 32);
    v16bf b1 = load_frag(brow + 32);
    #pragma unroll 2
    for (int k = 0; k < 960; k += 64) {
        __builtin_prefetch(brow + k + 320, 0, 1);
        v16bf a0n = load_frag(arow + k + 64);
        v16bf b0n = load_frag(brow + k + 64);
        v16bf a1n = load_frag(arow + k + 96);
        v16bf b1n = load_frag(brow + k + 96);
        acc0 = __builtin_amdgcn_wmma_f32_16x16x32_bf16(false, a0, false, b0, (short)0, acc0, false, false);
        acc1 = __builtin_amdgcn_wmma_f32_16x16x32_bf16(false, a1, false, b1, (short)0, acc1, false, false);
        a0 = a0n; b0 = b0n; a1 = a1n; b1 = b1n;
    }
    // epilogue
    acc0 = __builtin_amdgcn_wmma_f32_16x16x32_bf16(false, a0, false, b0, (short)0, acc0, false, false);
    acc1 = __builtin_amdgcn_wmma_f32_16x16x32_bf16(false, a1, false, b1, (short)0, acc1, false, false);

    const int col = gate * 1024 + hc * 16 + lo;
    #pragma unroll
    for (int i = 0; i < 8; ++i) {
        int ml = i + 8 * khi;
        int b_ = mt * 16 + ml;
        g_sm[gate][ml][lo] = acc0[i] + acc1[i] + xg[((size_t)b_ * SEQ + s) * G4 + col];
    }
    __syncthreads();

    for (int t = threadIdx.x; t < 256; t += 128) {
        int bl = t >> 4, nl = t & 15;
        int b_ = mt * 16 + bl;
        int hidx = hc * 16 + nl;
        float fg = sigm(g_sm[0][bl][nl]);
        float ig = sigm(g_sm[1][bl][nl]);
        float cn = tanhf(g_sm[2][bl][nl]);
        float og = sigm(g_sm[3][bl][nl]);
        size_t sidx = (size_t)b_ * HID + hidx;
        float cv = fg * c[sidx] + ig * cn;
        float hv = og * tanhf(cv);
        c[sidx] = cv;
        size_t oidx = ((size_t)b_ * SEQ + s) * HID + hidx;
        cs[oidx] = cv;
        hs[oidx] = hv;
        h_out[sidx] = f2bf(hv);
    }
}

extern "C" void kernel_launch(void* const* d_in, const int* in_sizes, int n_in,
                              void* d_out, int out_size, void* d_ws, size_t ws_size,
                              hipStream_t stream) {
    (void)in_sizes; (void)n_in; (void)out_size; (void)ws_size;
    const float* x   = (const float*)d_in[0];
    const float* Wf  = (const float*)d_in[1];
    const float* bf_ = (const float*)d_in[2];
    const float* Wi  = (const float*)d_in[3];
    const float* bi_ = (const float*)d_in[4];
    const float* Wc  = (const float*)d_in[5];
    const float* bc_ = (const float*)d_in[6];
    const float* Wo  = (const float*)d_in[7];
    const float* bo_ = (const float*)d_in[8];

    float* hs = (float*)d_out;                               // [B,S,H]
    float* cs = hs + (size_t)BATCH * SEQ * HID;              // [B,S,H]

    // workspace layout (byte offsets, all 256-aligned)
    char* ws = (char*)d_ws;
    bf16t* WxT   = (bf16t*)(ws + 0);                         //  8 MB
    bf16t* WhT   = (bf16t*)(ws + 8388608u);                  //  8 MB
    float* bias  = (float*)(ws + 16777216u);                 // 16 KB
    bf16t* xb    = (bf16t*)(ws + 16793600u);                 // 64 MB
    bf16t* hbuf0 = (bf16t*)(ws + 83902464u);                 // 128 KB
    bf16t* hbuf1 = (bf16t*)(ws + 84033536u);                 // 128 KB
    float* cbuf  = (float*)(ws + 84164608u);                 // 256 KB
    float* xg    = (float*)(ws + 84426752u);                 // 512 MB

    pack_weights<<<32768, 256, 0, stream>>>(Wf, Wi, Wc, Wo, WxT, WhT);
    pack_bias<<<16, 256, 0, stream>>>(bf_, bi_, bc_, bo_, bias);
    convert_x<<<32768, 256, 0, stream>>>(x, xb, BATCH * SEQ * INSZ);
    init_state<<<256, 256, 0, stream>>>(hbuf0, hbuf1, cbuf);

    gemm_xproj<<<16384, 256, 0, stream>>>(xb, WxT, bias, xg);

    for (int s = 0; s < SEQ; ++s) {
        const bf16t* hin = (s & 1) ? hbuf1 : hbuf0;
        bf16t*       hot = (s & 1) ? hbuf0 : hbuf1;
        lstm_step<<<dim3(64, 4), 128, 0, stream>>>(WhT, xg, hin, hot, cbuf, hs, cs, s);
    }
}
